// MultiHeadAttention_46858093199608
// MI455X (gfx1250) — compile-verified, matrix operand order checked by reading, not run on hardware
//
#include <hip/hip_runtime.h>
#include <cstdint>

// ---------------------------------------------------------------------------
// MI455X (gfx1250) fused causal multi-head attention, bf16 WMMA pipeline.
//   x[f32] -> bf16 -> Q/K/V proj (double-buffered async-LDS WMMA GEMM) ->
//   flash attention (WMMA + async LDS staging, online softmax) ->
//   output proj (WMMA GEMM, f32 out).
// ---------------------------------------------------------------------------

typedef __attribute__((ext_vector_type(16))) __bf16 v16bf;
typedef __attribute__((ext_vector_type(8)))  __bf16 v8bf;
typedef __attribute__((ext_vector_type(4)))  __bf16 v4bf;
typedef __attribute__((ext_vector_type(8)))  float  v8f;

// exact param type of __builtin_amdgcn_global_load_async_to_lds_b128
typedef int v4i_b128 __attribute__((vector_size(16)));
typedef __attribute__((address_space(1))) v4i_b128* gptr_b128;
typedef __attribute__((address_space(3))) v4i_b128* lptr_b128;

#define N_HEADS 12
#define EMB     768
#define HDIM    64
#define TSEQ    1024
#define BATCH   8
#define MROWS   (BATCH * TSEQ)   // 8192
#define KSTEP   32

#if defined(__AMDGCN__) && __has_builtin(__builtin_amdgcn_global_load_async_to_lds_b128) && __has_builtin(__builtin_amdgcn_s_wait_asynccnt)
#define HAVE_ASYNC 1
#else
#define HAVE_ASYNC 0
#endif

__device__ __forceinline__ v8f wmma_bf16(v16bf a, v16bf b, v8f c) {
  // (neg_a, A, neg_b, B, c_mod, C, reuse_a, reuse_b)
  return __builtin_amdgcn_wmma_f32_16x16x32_bf16(false, a, false, b,
                                                 (short)0, c, false, false);
}

// 16-byte global -> LDS copy; async (ASYNCcnt-tracked) when available.
__device__ __forceinline__ void cp_async16(const void* g, void* l) {
#if HAVE_ASYNC
  __builtin_amdgcn_global_load_async_to_lds_b128(
      (gptr_b128)(uintptr_t)g,
      (lptr_b128)(unsigned)(uintptr_t)l,
      0, 0);
#else
  *(v8bf*)l = *(const v8bf*)g;
#endif
}
__device__ __forceinline__ void cp_async_wait() {
#if HAVE_ASYNC
  __builtin_amdgcn_s_wait_asynccnt(0);
#endif
}

// --------------------------- f32 -> bf16 cast ------------------------------
__global__ void cvt_bf16_kernel(const float* __restrict__ in,
                                __bf16* __restrict__ out, int n) {
  int i = (blockIdx.x * blockDim.x + threadIdx.x) * 4;
  if (i < n) {
    float4 f = *(const float4*)(in + i);
    v4bf o;
    o[0] = (__bf16)f.x; o[1] = (__bf16)f.y;
    o[2] = (__bf16)f.z; o[3] = (__bf16)f.w;
    *(v4bf*)(out + i) = o;
  }
}

// --------------------- GEMM: out = A(MxK) * W(NxK)^T + bias ----------------
// Block tile 64M x 256N, 8 waves (2x4), double-buffered LDS k-steps of 32.
// mode 0: f32 row-major [M,N] output; mode 1: bf16 head-split [B,H,T,D].
__global__ __launch_bounds__(256)
void gemm_bias_wmma(const __bf16* __restrict__ A, const __bf16* __restrict__ W,
                    const float* __restrict__ bias, void* __restrict__ out,
                    int M, int N, int K, int mode) {
  __shared__ alignas(32) __bf16 asl[2][64 * KSTEP];    //  8 KB A tiles
  __shared__ alignas(32) __bf16 bsl[2][256 * KSTEP];   // 32 KB B tiles

  const int tid  = threadIdx.x;
  const int wave = tid >> 5, lane = tid & 31, half = lane >> 4, l16 = lane & 15;
  const int mb = blockIdx.x * 64;
  const int nb = blockIdx.y * 256;
  const int wm = (wave >> 2) * 32;   // wave row offset within block
  const int wn = (wave & 3) * 64;    // wave col offset within block

  // staging assignment: each thread owns one B row (64B) + half own A chunks
  const int brow = tid;
  const int arow = tid >> 1;
  const int akof = (tid & 1) * 16;
  const __bf16* gB = W + (size_t)(nb + brow) * K;
  const __bf16* gA = A + (size_t)(mb + arow) * K + akof;

  v8f acc[2][4];
  #pragma unroll
  for (int mi = 0; mi < 2; ++mi)
    #pragma unroll
    for (int s = 0; s < 4; ++s) acc[mi][s] = (v8f){};

  auto stage = [&](int buf, int ks) {
    const __bf16* gb = gB + ks * KSTEP;
    __bf16* lb = &bsl[buf][brow * KSTEP];
    cp_async16(gb,      lb);
    cp_async16(gb +  8, lb +  8);
    cp_async16(gb + 16, lb + 16);
    cp_async16(gb + 24, lb + 24);
    if (tid < 128) {
      const __bf16* ga = gA + ks * KSTEP;
      __bf16* la = &asl[buf][arow * KSTEP + akof];
      cp_async16(ga,     la);
      cp_async16(ga + 8, la + 8);
    }
  };

  const int nks = K / KSTEP;
  stage(0, 0);
  int buf = 0;
  for (int ks = 0; ks < nks; ++ks, buf ^= 1) {
    cp_async_wait();
    __syncthreads();                       // tile[buf] ready, tile[buf^1] free
    if (ks + 1 < nks) stage(buf ^ 1, ks + 1);

    v16bf af0 = *(const v16bf*)(&asl[buf][(wm + l16) * KSTEP + half * 16]);
    v16bf af1 = *(const v16bf*)(&asl[buf][(wm + 16 + l16) * KSTEP + half * 16]);
    #pragma unroll
    for (int s = 0; s < 4; ++s) {
      v16bf bfr = *(const v16bf*)(&bsl[buf][(wn + s * 16 + l16) * KSTEP + half * 16]);
      acc[0][s] = wmma_bf16(af0, bfr, acc[0][s]);
      acc[1][s] = wmma_bf16(af1, bfr, acc[1][s]);
    }
  }

  #pragma unroll
  for (int mi = 0; mi < 2; ++mi) {
    #pragma unroll
    for (int s = 0; s < 4; ++s) {
      #pragma unroll
      for (int r = 0; r < 8; ++r) {
        const int m = mb + wm + mi * 16 + r + 8 * half;  // C: row = vgpr+8*half
        const int n = nb + wn + s * 16 + l16;            //    col = lane%16
        const float v = acc[mi][s][r] + bias[n];
        if (mode == 0) {
          ((float*)out)[(size_t)m * N + n] = v;
        } else {
          const int b = m >> 10, t = m & 1023;           // T = 1024
          const int h = n >> 6,  d = n & 63;             // D = 64
          ((__bf16*)out)[(((size_t)b * N_HEADS + h) * TSEQ + t) * HDIM + d] =
              (__bf16)v;
        }
      }
    }
  }
}

// ------------------------- causal flash attention --------------------------
// grid = (T/128, H, B); 8 waves, each wave owns a 16-row query tile.
// K tile: async global->LDS double buffer. V tile: register prefetch +
// transpose-scatter into double-buffered LDS.
__global__ __launch_bounds__(256)
void flash_attn_kernel(const __bf16* __restrict__ Q, const __bf16* __restrict__ K,
                       const __bf16* __restrict__ V, __bf16* __restrict__ ctx) {
  __shared__ alignas(32) __bf16 kls[2][32 * 64];     // K tiles   [key][d]
  __shared__ alignas(32) __bf16 vtls[2][64 * 32];    // V^T tiles [d][key]
  __shared__ alignas(32) __bf16 pls[8 * 16 * 32];    // per-wave P scratch

  const int tid  = threadIdx.x;
  const int wave = tid >> 5, lane = tid & 31, half = lane >> 4, l16 = lane & 15;
  const int qb = blockIdx.x, h = blockIdx.y, b = blockIdx.z;
  const size_t bh = (size_t)b * N_HEADS + h;
  const __bf16* Qb = Q + bh * TSEQ * HDIM;
  const __bf16* Kb = K + bh * TSEQ * HDIM;
  const __bf16* Vb = V + bh * TSEQ * HDIM;

  const int qg = qb * 128 + wave * 16;
  const int tq = qg + l16;
  const v16bf qf0 = *(const v16bf*)(Qb + (size_t)tq * HDIM + half * 16);
  const v16bf qf1 = *(const v16bf*)(Qb + (size_t)tq * HDIM + 32 + half * 16);

  v8f o0 = {}, o1 = {}, o2 = {}, o3 = {};
  float mrow[8], lrow[8];
  #pragma unroll
  for (int r = 0; r < 8; ++r) { mrow[r] = -1e30f; lrow[r] = 0.f; }

  const int kr = tid >> 3;            // staging: key row 0..31
  const int c  = (tid & 7) * 8;       //          d offset (8 bf16 = 16B)

  // prologue: stage key-block 0
  cp_async16(Kb + (size_t)kr * HDIM + c, &kls[0][kr * 64 + c]);
  v8bf vreg = *(const v8bf*)(Vb + (size_t)kr * HDIM + c);

  const int nkb = (qb + 1) * 4;       // causal key-block limit
  int buf = 0;
  for (int kb = 0; kb < nkb; ++kb, buf ^= 1) {
    cp_async_wait();
    #pragma unroll
    for (int e = 0; e < 8; ++e) vtls[buf][(c + e) * 32 + kr] = vreg[e];
    __syncthreads();
    if (kb + 1 < nkb) {               // prefetch next tile under compute
      cp_async16(Kb + (size_t)((kb + 1) * 32 + kr) * HDIM + c,
                 &kls[buf ^ 1][kr * 64 + c]);
      vreg = *(const v8bf*)(Vb + (size_t)((kb + 1) * 32 + kr) * HDIM + c);
    }

    const int kbase = kb * 32;
    // S = Q K^T : K-dim = d (64) -> two WMMA k-steps per 16x16 tile
    v8f s0 = {}, s1 = {};
    s0 = wmma_bf16(qf0, *(const v16bf*)(&kls[buf][l16 * 64 + half * 16]), s0);
    s1 = wmma_bf16(qf0, *(const v16bf*)(&kls[buf][(16 + l16) * 64 + half * 16]), s1);
    s0 = wmma_bf16(qf1, *(const v16bf*)(&kls[buf][l16 * 64 + 32 + half * 16]), s0);
    s1 = wmma_bf16(qf1, *(const v16bf*)(&kls[buf][(16 + l16) * 64 + 32 + half * 16]), s1);

    // online softmax over the 32-key block (1/sqrt(64) = 0.125)
    float alpha[8];
    #pragma unroll
    for (int r = 0; r < 8; ++r) {
      const int m = r + 8 * half;
      const int qrow = qg + m;
      float a = s0[r] * 0.125f;
      float cc = s1[r] * 0.125f;
      if (kbase + l16 > qrow)      a  = -1e30f;      // causal mask
      if (kbase + 16 + l16 > qrow) cc = -1e30f;
      float mx = fmaxf(a, cc);                       // row reduce in 16-lane half
      mx = fmaxf(mx, __shfl_xor(mx, 1, 32));
      mx = fmaxf(mx, __shfl_xor(mx, 2, 32));
      mx = fmaxf(mx, __shfl_xor(mx, 4, 32));
      mx = fmaxf(mx, __shfl_xor(mx, 8, 32));
      const float mnew = fmaxf(mrow[r], mx);
      const float ea = __expf(a - mnew);
      const float ec = __expf(cc - mnew);
      float rs = ea + ec;
      rs += __shfl_xor(rs, 1, 32);
      rs += __shfl_xor(rs, 2, 32);
      rs += __shfl_xor(rs, 4, 32);
      rs += __shfl_xor(rs, 8, 32);
      alpha[r] = __expf(mrow[r] - mnew);
      lrow[r]  = lrow[r] * alpha[r] + rs;
      mrow[r]  = mnew;
      // stash P in LDS to re-read in WMMA A-layout (wave-local; LDS pipe
      // is in-order per wave, no barrier needed)
      pls[wave * 512 + m * 32 + l16]      = (__bf16)ea;
      pls[wave * 512 + m * 32 + 16 + l16] = (__bf16)ec;
    }
    #pragma unroll
    for (int r = 0; r < 8; ++r) {
      o0[r] *= alpha[r]; o1[r] *= alpha[r];
      o2[r] *= alpha[r]; o3[r] *= alpha[r];
    }

    // O += P V : K-dim = 32 keys -> one WMMA per 16-col d subtile
    v16bf pf = *(const v16bf*)(&pls[wave * 512 + l16 * 32 + half * 16]);
    o0 = wmma_bf16(pf, *(const v16bf*)(&vtls[buf][( 0 + l16) * 32 + half * 16]), o0);
    o1 = wmma_bf16(pf, *(const v16bf*)(&vtls[buf][(16 + l16) * 32 + half * 16]), o1);
    o2 = wmma_bf16(pf, *(const v16bf*)(&vtls[buf][(32 + l16) * 32 + half * 16]), o2);
    o3 = wmma_bf16(pf, *(const v16bf*)(&vtls[buf][(48 + l16) * 32 + half * 16]), o3);
  }

  // epilogue: normalize, write ctx as [B, T, E] bf16 (GEMM-ready)
  v8f oo[4] = {o0, o1, o2, o3};
  #pragma unroll
  for (int s = 0; s < 4; ++s) {
    #pragma unroll
    for (int r = 0; r < 8; ++r) {
      const int m = r + 8 * half;
      const int trow = qg + m;
      const int d = s * 16 + l16;
      const float v = oo[s][r] / lrow[r];
      ctx[((size_t)(b * TSEQ + trow)) * EMB + h * HDIM + d] = (__bf16)v;
    }
  }
}

// ---------------------------------------------------------------------------
extern "C" void kernel_launch(void* const* d_in, const int* in_sizes, int n_in,
                              void* d_out, int out_size, void* d_ws, size_t ws_size,
                              hipStream_t stream) {
  (void)in_sizes; (void)n_in; (void)out_size; (void)ws_size;
  const float* x  = (const float*)d_in[0];
  const float* Wq = (const float*)d_in[1];
  const float* bq = (const float*)d_in[2];
  const float* Wk = (const float*)d_in[3];
  const float* bk = (const float*)d_in[4];
  const float* Wv = (const float*)d_in[5];
  const float* bv = (const float*)d_in[6];
  const float* Wo = (const float*)d_in[7];
  const float* bo = (const float*)d_in[8];

  const size_t nX = (size_t)MROWS * EMB;   // 6,291,456
  const size_t nW = (size_t)EMB * EMB;     //   589,824
  __bf16* xb  = (__bf16*)d_ws;             // workspace carve (~67.6 MB bf16)
  __bf16* wqb = xb  + nX;
  __bf16* wkb = wqb + nW;
  __bf16* wvb = wkb + nW;
  __bf16* wob = wvb + nW;
  __bf16* Qh  = wob + nW;                  // [B,H,T,D]
  __bf16* Kh  = Qh  + nX;
  __bf16* Vh  = Kh  + nX;
  __bf16* ctx = Vh  + nX;                  // [B,T,E]

  cvt_bf16_kernel<<<(nX / 4 + 255) / 256, 256, 0, stream>>>(x,  xb,  (int)nX);
  cvt_bf16_kernel<<<(nW / 4 + 255) / 256, 256, 0, stream>>>(Wq, wqb, (int)nW);
  cvt_bf16_kernel<<<(nW / 4 + 255) / 256, 256, 0, stream>>>(Wk, wkb, (int)nW);
  cvt_bf16_kernel<<<(nW / 4 + 255) / 256, 256, 0, stream>>>(Wv, wvb, (int)nW);
  cvt_bf16_kernel<<<(nW / 4 + 255) / 256, 256, 0, stream>>>(Wo, wob, (int)nW);

  dim3 gg(MROWS / 64, EMB / 256);          // (128, 3)
  gemm_bias_wmma<<<gg, 256, 0, stream>>>(xb, wqb, bq, Qh, MROWS, EMB, EMB, 1);
  gemm_bias_wmma<<<gg, 256, 0, stream>>>(xb, wkb, bk, Kh, MROWS, EMB, EMB, 1);
  gemm_bias_wmma<<<gg, 256, 0, stream>>>(xb, wvb, bv, Vh, MROWS, EMB, EMB, 1);

  flash_attn_kernel<<<dim3(TSEQ / 128, N_HEADS, BATCH), 256, 0, stream>>>(
      Qh, Kh, Vh, ctx);

  gemm_bias_wmma<<<gg, 256, 0, stream>>>(ctx, wob, bo, d_out, MROWS, EMB, EMB, 0);
}
